// AdaptiveMultiScale_59030030516463
// MI455X (gfx1250) — compile-verified, matrix operand order checked by reading, not run on hardware
//
#include <hip/hip_runtime.h>

typedef __attribute__((ext_vector_type(16))) __bf16 v16bf;
typedef __attribute__((ext_vector_type(8)))  __bf16 v8bf;
typedef __attribute__((ext_vector_type(8)))  float  v8f;
typedef int v4i __attribute__((vector_size(16)));
typedef __attribute__((address_space(1))) v4i* gptr_v4i;
typedef __attribute__((address_space(3))) v4i* lptr_v4i;

static constexpr int cB = 64, cL = 192, cN = 32, cD = 128, cDff = 256, cE = 4;
static constexpr int cM = cB * cL * cN;           // 393216 tokens
static constexpr int cHD = 32, cH = cD / cHD;     // 4 heads of 32

#if __has_builtin(__builtin_amdgcn_global_load_async_to_lds_b128) && \
    __has_builtin(__builtin_amdgcn_s_wait_asynccnt)
#define USE_ASYNC_LDS 1
#else
#define USE_ASYNC_LDS 0
#endif

// ---------------- fragment layout helpers (CDNA5 WMMA 16-bit A/B) ----------
// B (32x16, KxN): lane = n + (k&8 ? 16:0); per-lane slot = (k>>4)*8 | (k&7)
// A (16x32, MxK): same mapping; note per lane the 16 slots are two contiguous
// 16-byte chunks of a row-major row: k in {0..7,16..23} (lanes 0-15) or
// {8..15,24..31} (lanes 16-31)  ->  row-major LDS + 2x b128 reads suffice.
__device__ __forceinline__ int frag_lane(int mn, int k) { return (mn & 15) + ((k & 8) ? 16 : 0); }
__device__ __forceinline__ int frag_slot(int k)         { return ((k >> 4) << 3) | (k & 7); }

__device__ __forceinline__ v16bf load_afrag(const __bf16* row, int lane) {
  const int off = (lane & 16) ? 8 : 0;
  v8bf lo = *(const v8bf*)(row + off);
  v8bf hi = *(const v8bf*)(row + off + 16);
  return __builtin_shufflevector(lo, hi, 0,1,2,3,4,5,6,7,8,9,10,11,12,13,14,15);
}

#if USE_ASYNC_LDS
__device__ __forceinline__ void async_copy16(const void* gp, void* lp) {
  __builtin_amdgcn_global_load_async_to_lds_b128(
      (gptr_v4i)(__attribute__((address_space(1))) void*)gp,
      (lptr_v4i)(__attribute__((address_space(3))) void*)lp, 0, 0);
}
#endif

// ---------------- generic WMMA GEMM --------------------------------------
// C[M, NT*16] = epilogue( A[M, KS*32] @ W[KS*32, NT*16] ), BM = 128*MT rows/block
// EPI 0: outB = bf16(relu? max(v,0):v)
// EPI 1: outF = (res? res : 0) + v
// EPI 2: outF += gates[row/rowsPerB, expert] * (res + v)   (final gated combine)
template <typename TA, int NT, int KS, int MT, bool RELU, int EPI>
__global__ __launch_bounds__(256) void gemm_wmma(
    const TA* __restrict__ A, const __bf16* __restrict__ W,
    __bf16* __restrict__ outB, float* __restrict__ outF,
    const float* __restrict__ res, const float* __restrict__ gates,
    int expertIdx, int rowsPerB)
{
  constexpr int KDIM = KS * 32;
  constexpr int NDIM = NT * 16;
  constexpr int BM   = 128 * MT;
  __shared__ __align__(32) __bf16 Wl[KS][NT][32][16];
  __shared__ __align__(64) __bf16 Al[2][BM][32];   // row-major, double buffered

  const int tid  = threadIdx.x;
  const int lane = tid & 31;
  const int wv   = tid >> 5;
  const size_t row0 = (size_t)blockIdx.x * BM;

  // stage weights in fragment order (resident for all K-steps)
  for (int idx = tid; idx < KDIM * NDIM; idx += 256) {
    int k = idx / NDIM, n = idx % NDIM;
    Wl[k >> 5][n >> 4][frag_lane(n, k & 31)][frag_slot(k & 31)] =
        W[(size_t)k * NDIM + n];
  }

  auto stageA = [&](int ks, int buf) {
    if constexpr (sizeof(TA) == 2) {
      // bf16 source: straight 16B copies (async DMA into LDS when available)
      for (int idx = tid; idx < BM * 4; idx += 256) {
        int r = idx >> 2, c = (idx & 3) * 8;
        const TA* gp = &A[(row0 + (size_t)r) * KDIM + ks * 32 + c];
        __bf16*   lp = &Al[buf][r][c];
#if USE_ASYNC_LDS
        async_copy16(gp, lp);
#else
        *(v8bf*)lp = *(const v8bf*)gp;
#endif
      }
    } else {
      // fp32 source: convert to bf16 while staging
      for (int idx = tid; idx < BM * 16; idx += 256) {
        int r = idx >> 4, c = (idx & 15) * 2;
        const float2 g = *(const float2*)&A[(row0 + (size_t)r) * KDIM + ks * 32 + c];
        Al[buf][r][c]     = (__bf16)g.x;
        Al[buf][r][c + 1] = (__bf16)g.y;
      }
      if (ks + 1 < KS)
        __builtin_prefetch(&A[(row0 + (size_t)(tid >> 1)) * KDIM + (ks + 1) * 32], 0, 0);
    }
  };

  v8f acc[MT][NT] = {};

  stageA(0, 0);
#if USE_ASYNC_LDS
  if constexpr (sizeof(TA) == 2) __builtin_amdgcn_s_wait_asynccnt(0);
#endif
  __syncthreads();

  for (int ks = 0; ks < KS; ++ks) {
    if (ks + 1 < KS) stageA(ks + 1, (ks + 1) & 1);  // overlap with WMMA below
    const int buf = ks & 1;
    v16bf a[MT];
#pragma unroll
    for (int mt = 0; mt < MT; ++mt)
      a[mt] = load_afrag(&Al[buf][(wv * MT + mt) * 16 + (lane & 15)][0], lane);
#pragma unroll
    for (int t = 0; t < NT; ++t) {
      v16bf b = *(const v16bf*)(&Wl[ks][t][lane][0]);
#pragma unroll
      for (int mt = 0; mt < MT; ++mt)
        acc[mt][t] = __builtin_amdgcn_wmma_f32_16x16x32_bf16(
            false, a[mt], false, b, (short)0, acc[mt][t], false, false);
    }
#if USE_ASYNC_LDS
    if constexpr (sizeof(TA) == 2) __builtin_amdgcn_s_wait_asynccnt(0);
#endif
    __syncthreads();
  }

  // D layout: vgpr r -> m = r + (lane>=16 ? 8:0), n = lane&15
  const int mhi = (lane >> 4) << 3;
#pragma unroll
  for (int mt = 0; mt < MT; ++mt) {
#pragma unroll
    for (int t = 0; t < NT; ++t) {
#pragma unroll
      for (int r = 0; r < 8; ++r) {
        size_t row = row0 + (size_t)((wv * MT + mt) * 16 + mhi + r);
        int    col = t * 16 + (lane & 15);
        float  v   = acc[mt][t][r];
        if constexpr (EPI == 0) {
          if constexpr (RELU) v = v > 0.f ? v : 0.f;
          outB[row * NDIM + col] = (__bf16)v;
        } else if constexpr (EPI == 1) {
          float rv = res ? res[row * NDIM + col] : 0.f;
          outF[row * NDIM + col] = rv + v;
        } else {
          float rv = res[row * NDIM + col];
          float g  = gates[(row / (size_t)rowsPerB) * cE + expertIdx];
          outF[row * NDIM + col] += g * (rv + v);
        }
      }
    }
  }
}

// ---------------- fused Q/K/V projection GEMM -----------------------------
// Reads A (fp32) once, produces three bf16 outputs: AI ~3x of split kernels.
__global__ __launch_bounds__(256) void gemm_qkv(
    const float* __restrict__ A,
    const __bf16* __restrict__ Wq, const __bf16* __restrict__ Wk,
    const __bf16* __restrict__ Wv,
    __bf16* __restrict__ Qo, __bf16* __restrict__ Ko, __bf16* __restrict__ Vo)
{
  constexpr int KS = 4, NT = 8, KDIM = 128, NDIM = 128;
  __shared__ __align__(32) __bf16 Wl[3][KS][NT][32][16];
  __shared__ __align__(64) __bf16 Al[2][128][32];

  const int tid  = threadIdx.x;
  const int lane = tid & 31;
  const int wv   = tid >> 5;
  const size_t row0 = (size_t)blockIdx.x * 128;

  for (int w = 0; w < 3; ++w) {
    const __bf16* W = (w == 0) ? Wq : (w == 1) ? Wk : Wv;
    for (int idx = tid; idx < KDIM * NDIM; idx += 256) {
      int k = idx >> 7, n = idx & 127;
      Wl[w][k >> 5][n >> 4][frag_lane(n, k & 31)][frag_slot(k & 31)] = W[idx];
    }
  }

  auto stageA = [&](int ks, int buf) {
    for (int idx = tid; idx < 128 * 16; idx += 256) {
      int r = idx >> 4, c = (idx & 15) * 2;
      const float2 g = *(const float2*)&A[(row0 + (size_t)r) * KDIM + ks * 32 + c];
      Al[buf][r][c]     = (__bf16)g.x;
      Al[buf][r][c + 1] = (__bf16)g.y;
    }
  };

  v8f accq[NT] = {}, acck[NT] = {}, accv[NT] = {};

  stageA(0, 0);
  __syncthreads();
  for (int ks = 0; ks < KS; ++ks) {
    if (ks + 1 < KS) stageA(ks + 1, (ks + 1) & 1);
    v16bf a = load_afrag(&Al[ks & 1][wv * 16 + (lane & 15)][0], lane);
#pragma unroll
    for (int t = 0; t < NT; ++t) {
      v16bf bq = *(const v16bf*)(&Wl[0][ks][t][lane][0]);
      accq[t] = __builtin_amdgcn_wmma_f32_16x16x32_bf16(false, a, false, bq, (short)0, accq[t], false, false);
      v16bf bk = *(const v16bf*)(&Wl[1][ks][t][lane][0]);
      acck[t] = __builtin_amdgcn_wmma_f32_16x16x32_bf16(false, a, false, bk, (short)0, acck[t], false, false);
      v16bf bv = *(const v16bf*)(&Wl[2][ks][t][lane][0]);
      accv[t] = __builtin_amdgcn_wmma_f32_16x16x32_bf16(false, a, false, bv, (short)0, accv[t], false, false);
    }
    __syncthreads();
  }

  const int mhi = (lane >> 4) << 3;
#pragma unroll
  for (int t = 0; t < NT; ++t) {
#pragma unroll
    for (int r = 0; r < 8; ++r) {
      size_t row = row0 + (size_t)(wv * 16 + mhi + r);
      size_t o   = row * NDIM + t * 16 + (lane & 15);
      Qo[o] = (__bf16)accq[t][r];
      Ko[o] = (__bf16)acck[t][r];
      Vo[o] = (__bf16)accv[t][r];
    }
  }
}

// ---------------- small attention core (intra & inter share layout) -------
// Sequences of length S (<=32), token stride N tokens: base = group*S*N + n
__global__ __launch_bounds__(128) void attn_kernel(
    const __bf16* __restrict__ Q, const __bf16* __restrict__ K,
    const __bf16* __restrict__ V, __bf16* __restrict__ O, int S)
{
  __shared__ float Ks[cH][32][cHD];
  __shared__ float Vs[cH][32][cHD];
  __shared__ float Ss[cH][32][32];
  const int h = threadIdx.x >> 5, lane = threadIdx.x & 31;
  const int n = blockIdx.x % cN;
  const size_t g = blockIdx.x / cN;
  const size_t base = g * (size_t)S * cN + n;

  for (int j = 0; j < S; ++j) {
    size_t tok = base + (size_t)j * cN;
    Ks[h][j][lane] = (float)K[tok * cD + h * cHD + lane];
    Vs[h][j][lane] = (float)V[tok * cD + h * cHD + lane];
  }
  __syncthreads();

  if (lane < S) {
    size_t tok = base + (size_t)lane * cN;
    const __bf16* qp = Q + tok * cD + h * cHD;
    float q[cHD];
#pragma unroll
    for (int d = 0; d < cHD; ++d) q[d] = (float)qp[d];

    float mx = -1e30f;
    for (int j = 0; j < S; ++j) {
      float s = 0.f;
#pragma unroll
      for (int d = 0; d < cHD; ++d) s += q[d] * Ks[h][j][d];
      s *= 0.17677669529663687f;  // 1/sqrt(32)
      Ss[h][lane][j] = s;
      mx = fmaxf(mx, s);
    }
    float sum = 0.f;
    for (int j = 0; j < S; ++j) { float e = __expf(Ss[h][lane][j] - mx); Ss[h][lane][j] = e; sum += e; }
    float inv = 1.f / sum;

    float o[cHD];
#pragma unroll
    for (int d = 0; d < cHD; ++d) o[d] = 0.f;
    for (int j = 0; j < S; ++j) {
      float a = Ss[h][lane][j] * inv;
#pragma unroll
      for (int d = 0; d < cHD; ++d) o[d] += a * Vs[h][j][d];
    }
    __bf16* op = O + tok * cD + h * cHD;
#pragma unroll
    for (int d = 0; d < cHD; ++d) op[d] = (__bf16)o[d];
  }
}

// ---------------- router: decompose (trend + top-3 Fourier season) --------
__global__ __launch_bounds__(192) void router_decompose(
    const float* __restrict__ x, float* __restrict__ newx)
{
  __shared__ float xs[cL];
  __shared__ float re[97], im[97], amp[97];
  __shared__ int keep[3];
  const int l = threadIdx.x;
  const int b = blockIdx.x / cN, n = blockIdx.x % cN;

  xs[l] = x[(((size_t)b * cL + l) * cN + n) * cD];  // channel 0
  __syncthreads();

  if (l < 97) {
    float sr = 0.f, si = 0.f;
    for (int t = 0; t < cL; ++t) {
      float ang = -6.2831853071795864f * (float)(l * t) / (float)cL;
      sr += xs[t] * __cosf(ang);
      si += xs[t] * __sinf(ang);
    }
    re[l] = sr; im[l] = si;
    amp[l] = (l == 0) ? 0.f : sqrtf(sr * sr + si * si);  // DC dropped
  }
  __syncthreads();
  if (l == 0) {
    for (int t = 0; t < 3; ++t) {
      int bi = 1; float bv = -1.f;
      for (int f = 1; f < 97; ++f) if (amp[f] > bv) { bv = amp[f]; bi = f; }
      keep[t] = bi; amp[bi] = -1.f;
    }
  }
  __syncthreads();

  float season = 0.f;
#pragma unroll
  for (int t = 0; t < 3; ++t) {
    int f = keep[t];
    float ang = 6.2831853071795864f * (float)(f * l) / (float)cL;
    float fac = (f == 0 || f == 96) ? 1.f : 2.f;
    season += (fac / (float)cL) * (re[f] * __cosf(ang) - im[f] * __sinf(ang));
  }
  float trend = 0.f;
  const int kers[3] = {4, 8, 12};
#pragma unroll
  for (int t = 0; t < 3; ++t) {
    int k = kers[t], lp = (k - 1) / 2;
    float s = 0.f;
    for (int j = 0; j < 12; ++j) {
      if (j >= k) break;
      int ii = l - lp + j; ii = ii < 0 ? 0 : (ii > cL - 1 ? cL - 1 : ii);
      s += xs[ii];
    }
    trend += s / (float)k;
  }
  trend *= (1.f / 3.f);
  newx[((size_t)b * cL + l) * cN + n] = xs[l] + season + trend;
}

// ---------------- router: h = newx . start_w ; logits ; top-2 gates -------
__global__ __launch_bounds__(192) void router_gates(
    const float* __restrict__ newx, const float* __restrict__ start_w,
    const float* __restrict__ gate_w, float* __restrict__ gates)
{
  __shared__ float hs[cL];
  __shared__ float logit[cE];
  const int l = threadIdx.x, b = blockIdx.x;
  float hv = 0.f;
  for (int n = 0; n < cN; ++n)
    hv += newx[((size_t)b * cL + l) * cN + n] * start_w[n];
  hs[l] = hv;
  __syncthreads();
  if (l < cE) {
    float s = 0.f;
    for (int t = 0; t < cL; ++t) s += hs[t] * gate_w[t * cE + l];
    logit[l] = s;
  }
  __syncthreads();
  if (l == 0) {
    int i0 = 0;
    for (int e = 1; e < cE; ++e) if (logit[e] > logit[i0]) i0 = e;
    int i1 = -1;
    for (int e = 0; e < cE; ++e)
      if (e != i0 && (i1 < 0 || logit[e] > logit[i1])) i1 = e;
    float m = logit[i0];
    float e0 = __expf(logit[i0] - m), e1 = __expf(logit[i1] - m);
    float inv = 1.f / (e0 + e1);
    for (int e = 0; e < cE; ++e) gates[b * cE + e] = 0.f;
    gates[b * cE + i0] = e0 * inv;
    gates[b * cE + i1] = e1 * inv;
  }
}

__global__ void balance_loss_kernel(const float* __restrict__ gates,
                                    float* __restrict__ out, size_t lossIdx)
{
  if (threadIdx.x != 0 || blockIdx.x != 0) return;
  float imp[cE] = {}, ld[cE] = {};
  for (int b = 0; b < cB; ++b)
#pragma unroll
    for (int e = 0; e < cE; ++e) {
      float g = gates[b * cE + e];
      imp[e] += g;
      ld[e] += (g > 0.f) ? 1.f : 0.f;
    }
  float loss = 0.f;
  for (int which = 0; which < 2; ++which) {
    const float* v = which ? ld : imp;
    float mean = 0.f;
    for (int e = 0; e < cE; ++e) mean += v[e];
    mean *= (1.f / cE);
    float var = 0.f;
    for (int e = 0; e < cE; ++e) { float d = v[e] - mean; var += d * d; }
    var *= (1.f / (cE - 1));  // ddof=1
    loss += var / (mean * mean + 1e-10f);
  }
  out[lossIdx] = loss * 0.01f;
}

// ---------------- misc elementwise kernels --------------------------------
__global__ void init_out_kernel(const float* __restrict__ x, float* __restrict__ out) {
  size_t i = (size_t)blockIdx.x * 256 + threadIdx.x;
  out[i] = x[i];
}
__global__ void cast_bf16_kernel(const float* __restrict__ s, __bf16* __restrict__ d, int n) {
  int i = blockIdx.x * 256 + threadIdx.x;
  if (i < n) d[i] = (__bf16)s[i];
}
__global__ void pool_kernel(const float* __restrict__ x1, float* __restrict__ u, int P) {
  size_t idx = (size_t)blockIdx.x * 256 + threadIdx.x;  // over Mp*D
  size_t t = idx >> 7;  int d = (int)(idx & 127);
  size_t g = t / cN;    int n = (int)(t % cN);
  size_t base = ((g * P) * (size_t)cN + n) * cD + d;
  float s = 0.f;
  for (int p = 0; p < P; ++p) s += x1[base + (size_t)p * cN * cD];
  u[idx] = s / (float)P;
}
__global__ void x2build_kernel(const float* __restrict__ x1, const float* __restrict__ o2,
                               float* __restrict__ x2, int P) {
  size_t idx = (size_t)blockIdx.x * 256 + threadIdx.x;  // over M*D
  size_t t = idx >> 7;  int d = (int)(idx & 127);
  size_t gp = t / cN;   int n = (int)(t % cN);
  size_t g = gp / P;
  x2[idx] = x1[idx] + o2[(g * cN + n) * (size_t)cD + d];
}

// ---------------- host-side orchestration ---------------------------------
extern "C" void kernel_launch(void* const* d_in, const int* in_sizes, int n_in,
                              void* d_out, int out_size, void* d_ws, size_t ws_size,
                              hipStream_t stream) {
  const float* x       = (const float*)d_in[0];
  const float* start_w = (const float*)d_in[1];
  const float* gate_w  = (const float*)d_in[2];
  const float* Wsrc[10];
  for (int i = 0; i < 10; ++i) Wsrc[i] = (const float*)d_in[3 + i];
  float* out = (float*)d_out;

  char* p = (char*)d_ws;
  auto alloc = [&](size_t bytes) -> void* {
    void* r = (void*)p;
    p += (bytes + 255) & ~(size_t)255;
    return r;
  };
  const size_t DD = (size_t)cD * cD, DF = (size_t)cD * cDff;
  const int MpMax = cB * (cL / 6) * cN;  // 65536

  float*  gates = (float*)alloc((size_t)cB * cE * 4);
  float*  newx  = (float*)alloc((size_t)cB * cL * cN * 4);
  __bf16* Wb[10];
  int wElems[10] = {(int)(cE*DD),(int)(cE*DD),(int)(cE*DD),(int)(cE*DD),
                    (int)(cE*DD),(int)(cE*DD),(int)(cE*DD),(int)(cE*DD),
                    (int)(cE*DF),(int)(cE*DF)};
  for (int i = 0; i < 10; ++i) Wb[i] = (__bf16*)alloc((size_t)wElems[i] * 2);
  __bf16* Qb  = (__bf16*)alloc((size_t)cM * cD * 2);
  __bf16* Kb  = (__bf16*)alloc((size_t)cM * cD * 2);
  __bf16* Vb  = (__bf16*)alloc((size_t)cM * cD * 2);
  __bf16* Ob  = (__bf16*)alloc((size_t)cM * cD * 2);
  float*  x1  = (float*) alloc((size_t)cM * cD * 4);
  float*  x2  = (float*) alloc((size_t)cM * cD * 4);
  float*  u   = (float*) alloc((size_t)MpMax * cD * 4);
  __bf16* Q2b = (__bf16*)alloc((size_t)MpMax * cD * 2);
  __bf16* K2b = (__bf16*)alloc((size_t)MpMax * cD * 2);
  __bf16* V2b = (__bf16*)alloc((size_t)MpMax * cD * 2);
  __bf16* O2b = (__bf16*)alloc((size_t)MpMax * cD * 2);
  float*  o2f = (float*) alloc((size_t)MpMax * cD * 4);
  __bf16* hff = (__bf16*)alloc((size_t)cM * cDff * 2);

  for (int i = 0; i < 10; ++i)
    cast_bf16_kernel<<<(wElems[i] + 255) / 256, 256, 0, stream>>>(Wsrc[i], Wb[i], wElems[i]);

  router_decompose<<<cB * cN, 192, 0, stream>>>(x, newx);
  router_gates<<<cB, 192, 0, stream>>>(newx, start_w, gate_w, gates);
  balance_loss_kernel<<<1, 32, 0, stream>>>(gates, out, (size_t)out_size - 1);
  init_out_kernel<<<((size_t)cM * cD) / 256, 256, 0, stream>>>(x, out);

  const int PS[cE] = {24, 12, 8, 6};
  for (int e = 0; e < cE; ++e) {
    const int P = PS[e], Pn = cL / P, Mp = cB * Pn * cN;
    const dim3 blk(256);

    // intra-patch: fused Q/K/V projection, attention, x1 = x + attn@Wo
    gemm_qkv<<<cM / 128, blk, 0, stream>>>(x, Wb[0] + e * DD, Wb[1] + e * DD,
                                           Wb[2] + e * DD, Qb, Kb, Vb);
    attn_kernel<<<cB * Pn * cN, 128, 0, stream>>>(Qb, Kb, Vb, Ob, P);
    gemm_wmma<__bf16, 8, 4, 2, false, 1><<<cM / 256, blk, 0, stream>>>(
        Ob, Wb[3] + e * DD, nullptr, x1, x, nullptr, 0, 0);

    // inter-patch: pool, fused Q2/K2/V2, attention over Pn, o2, broadcast add
    pool_kernel<<<((size_t)Mp * cD) / 256, 256, 0, stream>>>(x1, u, P);
    gemm_qkv<<<Mp / 128, blk, 0, stream>>>(u, Wb[4] + e * DD, Wb[5] + e * DD,
                                           Wb[6] + e * DD, Q2b, K2b, V2b);
    attn_kernel<<<cB * cN, 128, 0, stream>>>(Q2b, K2b, V2b, O2b, Pn);
    gemm_wmma<__bf16, 8, 4, 2, false, 1><<<Mp / 256, blk, 0, stream>>>(
        O2b, Wb[7] + e * DD, nullptr, o2f, nullptr, nullptr, 0, 0);
    x2build_kernel<<<((size_t)cM * cD) / 256, 256, 0, stream>>>(x1, o2f, x2, P);

    // FFN + fused gated combine: out += gate * (x2 + relu(x2@W1)@W2)
    gemm_wmma<float, 16, 4, 1, true, 0><<<cM / 128, blk, 0, stream>>>(
        x2, Wb[8] + e * DF, hff, nullptr, nullptr, nullptr, 0, 0);
    gemm_wmma<__bf16, 8, 8, 2, false, 2><<<cM / 256, blk, 0, stream>>>(
        hff, Wb[9] + e * DF, nullptr, out, x2, gates, e, cL * cN);
  }
}
